// SelectiveSSM_61065845015401
// MI455X (gfx1250) — matrix operand order repaired and checked by reading, not compiled, not run on hardware
//
#include <hip/hip_runtime.h>
#include <hip/hip_bf16.h>
#include <math.h>

#define B_SZ    2
#define L_SZ    2048
#define DM      1024
#define N_ST    16
#define RK      64
#define NX      96          // RK + 2*N_ST
#define TOK     (B_SZ*L_SZ) // 4096
#define LC      128         // chunk length
#define NC      (L_SZ/LC)   // 16 chunks

typedef __attribute__((ext_vector_type(16))) __bf16 v16bf;
typedef __attribute__((ext_vector_type(8)))  float  v8f;

__device__ __forceinline__ void split_bf16(float v, __bf16& hi, __bf16& lo) {
    hi = (__bf16)v;
    lo = (__bf16)(v - (float)hi);
}

// ---------------------------------------------------------------------------
// Prep: transpose + bf16-split weights into B-operand friendly [n][k] layout.
// ---------------------------------------------------------------------------
__global__ void prep_wx(const float* __restrict__ Wx,
                        unsigned short* __restrict__ hi_, unsigned short* __restrict__ lo_) {
    int idx = blockIdx.x * 256 + threadIdx.x;
    if (idx >= NX * DM) return;
    int n = idx / DM, k = idx % DM;
    __bf16 h, l; split_bf16(Wx[k * NX + n], h, l);
    ((__bf16*)hi_)[idx] = h; ((__bf16*)lo_)[idx] = l;
}

__global__ void prep_wdt(const float* __restrict__ Wdt,
                         unsigned short* __restrict__ hi_, unsigned short* __restrict__ lo_) {
    int idx = blockIdx.x * 256 + threadIdx.x;
    if (idx >= DM * RK) return;
    int n = idx / RK, k = idx % RK;
    __bf16 h, l; split_bf16(Wdt[k * DM + n], h, l);
    ((__bf16*)hi_)[idx] = h; ((__bf16*)lo_)[idx] = l;
}

// ---------------------------------------------------------------------------
// GEMM1: xp[4096,96] = x[4096,1024] @ W_x  (bf16x3 WMMA, fp32 accum)
// one 16x16 tile per wave, 8 waves/block
// ---------------------------------------------------------------------------
__global__ __launch_bounds__(256) void gemm_xp(const float* __restrict__ x,
                                               const unsigned short* __restrict__ Bhi,
                                               const unsigned short* __restrict__ Blo,
                                               float* __restrict__ xp) {
    const int lane = threadIdx.x & 31;
    const int wave = threadIdx.x >> 5;
    const int mt = blockIdx.x * 8 + wave;
    const int nt = blockIdx.y;
    const int rowA = mt * 16 + (lane & 15);
    const int kA   = (lane >> 4) * 16;   // A: half-wave K split {0..15}/{16..31}
    const int col  = nt * 16 + (lane & 15);
    const int kB   = (lane >> 4) * 16;   // B: half-wave K split {0..15}/{16..31}
    v8f acc = {};
    for (int k0 = 0; k0 < DM; k0 += 32) {
        const float* ap = x + rowA * DM + k0 + kA;
        v16bf ah, al;
#pragma unroll
        for (int i = 0; i < 8; i++) { __bf16 h, l; split_bf16(ap[i],      h, l); ah[i]   = h; al[i]   = l; }
#pragma unroll
        for (int i = 0; i < 8; i++) { __bf16 h, l; split_bf16(ap[8 + i], h, l); ah[8+i] = h; al[8+i] = l; }
        const v16bf bh = *(const v16bf*)(Bhi + col * DM + k0 + kB);
        const v16bf bl = *(const v16bf*)(Blo + col * DM + k0 + kB);
        acc = __builtin_amdgcn_wmma_f32_16x16x32_bf16(false, ah, false, bh, (short)0, acc, false, false);
        acc = __builtin_amdgcn_wmma_f32_16x16x32_bf16(false, al, false, bh, (short)0, acc, false, false);
        acc = __builtin_amdgcn_wmma_f32_16x16x32_bf16(false, ah, false, bl, (short)0, acc, false, false);
    }
#pragma unroll
    for (int j = 0; j < 8; j++) {
        int r = mt * 16 + (lane >> 4) * 8 + j;
        xp[r * NX + nt * 16 + (lane & 15)] = acc[j];
    }
}

// ---------------------------------------------------------------------------
// GEMM2 + softplus: dt[4096,1024] = softplus(xp[:, :64] @ W_dt + b_dt)
// ---------------------------------------------------------------------------
__global__ __launch_bounds__(256) void gemm_dt(const float* __restrict__ xp,
                                               const unsigned short* __restrict__ Bhi,
                                               const unsigned short* __restrict__ Blo,
                                               const float* __restrict__ bias,
                                               float* __restrict__ dt) {
    const int lane = threadIdx.x & 31;
    const int wave = threadIdx.x >> 5;
    const int mt = blockIdx.x * 8 + wave;
    const int nt = blockIdx.y;
    const int rowA = mt * 16 + (lane & 15);
    const int kA   = (lane >> 4) * 16;
    const int col  = nt * 16 + (lane & 15);
    const int kB   = (lane >> 4) * 16;
    v8f acc = {};
#pragma unroll
    for (int k0 = 0; k0 < RK; k0 += 32) {
        const float* ap = xp + rowA * NX + k0 + kA;
        v16bf ah, al;
#pragma unroll
        for (int i = 0; i < 8; i++) { __bf16 h, l; split_bf16(ap[i],      h, l); ah[i]   = h; al[i]   = l; }
#pragma unroll
        for (int i = 0; i < 8; i++) { __bf16 h, l; split_bf16(ap[8 + i], h, l); ah[8+i] = h; al[8+i] = l; }
        const v16bf bh = *(const v16bf*)(Bhi + col * RK + k0 + kB);
        const v16bf bl = *(const v16bf*)(Blo + col * RK + k0 + kB);
        acc = __builtin_amdgcn_wmma_f32_16x16x32_bf16(false, ah, false, bh, (short)0, acc, false, false);
        acc = __builtin_amdgcn_wmma_f32_16x16x32_bf16(false, al, false, bh, (short)0, acc, false, false);
        acc = __builtin_amdgcn_wmma_f32_16x16x32_bf16(false, ah, false, bl, (short)0, acc, false, false);
    }
    const float bz = bias[col];
#pragma unroll
    for (int j = 0; j < 8; j++) {
        int r = mt * 16 + (lane >> 4) * 8 + j;
        float z = acc[j] + bz;
        float s = (z > 20.f) ? z : log1pf(__expf(z));   // softplus
        dt[r * DM + col] = s;
    }
}

// ---------------------------------------------------------------------------
// Scan pass 1: per-chunk decay product P = prod(dA) and local end state (h0=0).
// block = 16 d-channels x 16 states; lane&15 = n (state index).
// ---------------------------------------------------------------------------
__global__ __launch_bounds__(256) void scan_pass1(const float* __restrict__ x,
                                                  const float* __restrict__ xp,
                                                  const float* __restrict__ dt,
                                                  const float* __restrict__ A_log,
                                                  float* __restrict__ P,
                                                  float* __restrict__ Hl) {
    __shared__ float sdt[LC * 16], sx[LC * 16], sB[LC * 16];
    const int tid = threadIdx.x;
    const int n  = tid & 15;
    const int dl = tid >> 4;
    const int d0 = blockIdx.x * 16;
    const int c  = blockIdx.y;
    const int b  = blockIdx.z;
    const int tok0 = b * L_SZ + c * LC;
    const int d = d0 + dl;

    for (int i = tid; i < LC * 16; i += 256) {
        int t = i >> 4, j = i & 15;
        sdt[i] = dt[(tok0 + t) * DM + d0 + j];
        sx[i]  = x [(tok0 + t) * DM + d0 + j];
        sB[i]  = xp[(tok0 + t) * NX + RK + j];
    }
    __syncthreads();

    const float Adn = -__expf(A_log[d * N_ST + n]);
    float h = 0.f, Pp = 1.f;
#pragma unroll 4
    for (int t = 0; t < LC; t++) {
        float dtv = sdt[t * 16 + dl];
        float a   = __expf(dtv * Adn);
        h  = a * h + dtv * sB[t * 16 + n] * sx[t * 16 + dl];
        Pp *= a;
    }
    size_t base = ((size_t)(b * NC + c) * DM + d) * N_ST + n;
    P[base]  = Pp;
    Hl[base] = h;
}

// ---------------------------------------------------------------------------
// Combine: 16-step serial scan over chunks -> carry-in state per chunk.
// one thread per (b,d,n); fully coalesced.
// ---------------------------------------------------------------------------
__global__ void scan_combine(const float* __restrict__ P, const float* __restrict__ Hl,
                             float* __restrict__ Hin) {
    int flat = blockIdx.x * 256 + threadIdx.x;          // < B*DM*N_ST = 32768
    if (flat >= B_SZ * DM * N_ST) return;
    int b   = flat >> 14;                               // / (DM*N_ST)
    int rem = flat & 16383;
    float h = 0.f;
#pragma unroll
    for (int c = 0; c < NC; c++) {
        size_t idx = (size_t)(b * NC + c) * (DM * N_ST) + rem;
        Hin[idx] = h;
        h = P[idx] * h + Hl[idx];
    }
}

// ---------------------------------------------------------------------------
// Scan pass 2: re-run chunk scan from carry-in, emit y = h.C + x*D.
// y reduction over n = 4-step shfl_xor butterfly within 16-lane groups (wave32).
// ---------------------------------------------------------------------------
__global__ __launch_bounds__(256) void scan_pass2(const float* __restrict__ x,
                                                  const float* __restrict__ xp,
                                                  const float* __restrict__ dt,
                                                  const float* __restrict__ A_log,
                                                  const float* __restrict__ Dv,
                                                  const float* __restrict__ Hin,
                                                  float* __restrict__ y) {
    __shared__ float sdt[LC * 16], sx[LC * 16], sB[LC * 16], sC[LC * 16];
    const int tid = threadIdx.x;
    const int n  = tid & 15;
    const int dl = tid >> 4;
    const int d0 = blockIdx.x * 16;
    const int c  = blockIdx.y;
    const int b  = blockIdx.z;
    const int tok0 = b * L_SZ + c * LC;
    const int d = d0 + dl;

    for (int i = tid; i < LC * 16; i += 256) {
        int t = i >> 4, j = i & 15;
        sdt[i] = dt[(tok0 + t) * DM + d0 + j];
        sx[i]  = x [(tok0 + t) * DM + d0 + j];
        sB[i]  = xp[(tok0 + t) * NX + RK + j];
        sC[i]  = xp[(tok0 + t) * NX + RK + N_ST + j];
    }
    __syncthreads();

    const float Adn = -__expf(A_log[d * N_ST + n]);
    const float Dd  = Dv[d];
    float h = Hin[((size_t)(b * NC + c) * DM + d) * N_ST + n];
#pragma unroll 2
    for (int t = 0; t < LC; t++) {
        float dtv = sdt[t * 16 + dl];
        float a   = __expf(dtv * Adn);
        h = a * h + dtv * sB[t * 16 + n] * sx[t * 16 + dl];
        float p = h * sC[t * 16 + n];
        p += __shfl_xor(p, 1, 32);
        p += __shfl_xor(p, 2, 32);
        p += __shfl_xor(p, 4, 32);
        p += __shfl_xor(p, 8, 32);
        if (n == 0)
            y[(size_t)(tok0 + t) * DM + d] = p + sx[t * 16 + dl] * Dd;
    }
}

// ---------------------------------------------------------------------------
extern "C" void kernel_launch(void* const* d_in, const int* in_sizes, int n_in,
                              void* d_out, int out_size, void* d_ws, size_t ws_size,
                              hipStream_t stream) {
    (void)in_sizes; (void)n_in; (void)out_size; (void)ws_size;
    const float* x    = (const float*)d_in[0];
    const float* Wx   = (const float*)d_in[1];
    const float* Wdt  = (const float*)d_in[2];
    const float* bdt  = (const float*)d_in[3];
    const float* Alog = (const float*)d_in[4];
    const float* Dv   = (const float*)d_in[5];
    float* y = (float*)d_out;

    char* ws = (char*)d_ws;
    size_t off = 0;
    float* xp  = (float*)(ws + off); off += (size_t)TOK * NX * sizeof(float);
    float* dt  = (float*)(ws + off); off += (size_t)TOK * DM * sizeof(float);
    float* P   = (float*)(ws + off); off += (size_t)B_SZ * NC * DM * N_ST * sizeof(float);
    float* Hl  = (float*)(ws + off); off += (size_t)B_SZ * NC * DM * N_ST * sizeof(float);
    float* Hin = (float*)(ws + off); off += (size_t)B_SZ * NC * DM * N_ST * sizeof(float);
    unsigned short* WxHi  = (unsigned short*)(ws + off); off += (size_t)NX * DM * 2;
    unsigned short* WxLo  = (unsigned short*)(ws + off); off += (size_t)NX * DM * 2;
    unsigned short* WdtHi = (unsigned short*)(ws + off); off += (size_t)DM * RK * 2;
    unsigned short* WdtLo = (unsigned short*)(ws + off); off += (size_t)DM * RK * 2;

    prep_wx <<<(NX * DM + 255) / 256, 256, 0, stream>>>(Wx,  WxHi,  WxLo);
    prep_wdt<<<(DM * RK + 255) / 256, 256, 0, stream>>>(Wdt, WdtHi, WdtLo);

    gemm_xp<<<dim3(TOK / 16 / 8, NX / 16), 256, 0, stream>>>(x, WxHi, WxLo, xp);
    gemm_dt<<<dim3(TOK / 16 / 8, DM / 16), 256, 0, stream>>>(xp, WdtHi, WdtLo, bdt, dt);

    scan_pass1 <<<dim3(DM / 16, NC, B_SZ), 256, 0, stream>>>(x, xp, dt, Alog, P, Hl);
    scan_combine<<<(B_SZ * DM * N_ST + 255) / 256, 256, 0, stream>>>(P, Hl, Hin);
    scan_pass2 <<<dim3(DM / 16, NC, B_SZ), 256, 0, stream>>>(x, xp, dt, Alog, Dv, Hin, y);
}
